// ClusterISAAttention_5471788335921
// MI455X (gfx1250) — compile-verified, matrix operand order checked by reading, not run on hardware
//
#include <hip/hip_runtime.h>

typedef __attribute__((ext_vector_type(16))) _Float16 v16h;
typedef __attribute__((ext_vector_type(8)))  float    v8f;

#define BB 2
#define NN 8192
#define WW 361
#define KK 49
#define CC 256
#define HH 8
#define HD 32

// Load 16 f16 fragment halves for one lane from a K-contiguous f32 row.
// Per CDNA5 ISA 7.12.2 (16-bit A 16x32): halves e=0..7 hold k = 8*khalf + e,
// halves e=8..15 hold k = 16 + 8*khalf + (e-8).  Caller passes p = row + kt*32 + 8*khalf.
__device__ inline v16h load_frag_f16(const float* __restrict__ p) {
    const float4 f0 = *(const float4*)(p);
    const float4 f1 = *(const float4*)(p + 4);
    const float4 f2 = *(const float4*)(p + 16);
    const float4 f3 = *(const float4*)(p + 20);
    v16h v;
    v[0]  = (_Float16)f0.x; v[1]  = (_Float16)f0.y; v[2]  = (_Float16)f0.z; v[3]  = (_Float16)f0.w;
    v[4]  = (_Float16)f1.x; v[5]  = (_Float16)f1.y; v[6]  = (_Float16)f1.z; v[7]  = (_Float16)f1.w;
    v[8]  = (_Float16)f2.x; v[9]  = (_Float16)f2.y; v[10] = (_Float16)f2.z; v[11] = (_Float16)f2.w;
    v[12] = (_Float16)f3.x; v[13] = (_Float16)f3.y; v[14] = (_Float16)f3.z; v[15] = (_Float16)f3.w;
    return v;
}

// C[M,256] = (A[M,256] @ W[256,256]^T + bias) * alpha   (torch Linear convention)
// One 16x64 output strip per wave: per 32-wide K step, 1 A fragment is reused
// against 4 B fragments -> 4x v_wmma_f32_16x16x32_f16 (32 wmma total).
__global__ __launch_bounds__(256) void gemm16_f16(const float* __restrict__ A,
                                                  const float* __restrict__ Wt,
                                                  const float* __restrict__ bias,
                                                  float* __restrict__ C,
                                                  int M, float alpha)
{
    const int wave   = threadIdx.x >> 5;
    const int lane   = threadIdx.x & 31;
    const int tilesM = (M + 15) >> 4;
    const int tid    = blockIdx.x * 8 + wave;         // wave-uniform
    if (tid >= tilesM * 4) return;                    // 256/64 == 4 strips per M-row
    const int tm    = tid >> 2;
    const int ts    = tid & 3;                        // which 64-wide N strip
    const int l15   = lane & 15;
    const int khalf = lane >> 4;

    const int arow = tm * 16 + l15;
    // Clamp instead of branch: garbage A rows only affect D rows we never store.
    const int arc  = (arow < M) ? arow : (M - 1);
    const float* ap = A  + (size_t)arc * CC;
    const float* bp = Wt + (size_t)(ts * 64 + l15) * CC;  // B[k][n] = W[n][k]

    v8f acc[4] = {{}, {}, {}, {}};
    #pragma unroll
    for (int kt = 0; kt < CC / 32; ++kt) {
        const int kb = kt * 32 + khalf * 8;
        const v16h av = load_frag_f16(ap + kb);
        #pragma unroll
        for (int sub = 0; sub < 4; ++sub) {
            const v16h bv = load_frag_f16(bp + (size_t)sub * 16 * CC + kb);
            acc[sub] = __builtin_amdgcn_wmma_f32_16x16x32_f16(false, av, false, bv,
                                                              (short)0, acc[sub],
                                                              false, false);
        }
    }

    #pragma unroll
    for (int sub = 0; sub < 4; ++sub) {
        const int col = ts * 64 + sub * 16 + l15;
        const float bb = bias[col];
        #pragma unroll
        for (int r = 0; r < 8; ++r) {
            const int row = tm * 16 + khalf * 8 + r;  // D layout: lanes16-31 hold M=r+8
            if (row < M)
                C[(size_t)row * CC + col] = (acc[sub][r] + bb) * alpha;
        }
    }
}

// One wave per (query, head): 49-key windowed softmax attention.
__global__ __launch_bounds__(256) void win_attn(const float* __restrict__ q,
                                                const float* __restrict__ kp,
                                                const float* __restrict__ vp,
                                                const int* __restrict__ idx,
                                                float* __restrict__ ao)
{
    const int bn   = blockIdx.x;                      // 0 .. B*N-1
    const int b    = bn / NN;
    const int h    = threadIdx.x >> 5;                // 8 waves = 8 heads
    const int lane = threadIdx.x & 31;
    const int w    = idx[bn];

    const float* qh = q + (size_t)bn * CC + h * HD;
    const size_t base = ((size_t)(b * WW + w) * KK) * CC + (size_t)h * HD;
    const float* kh = kp + base;
    const float* vh = vp + base;

    float qv[HD];
    #pragma unroll
    for (int d = 0; d < HD; d += 4) {
        const float4 t = *(const float4*)(qh + d);
        qv[d] = t.x; qv[d+1] = t.y; qv[d+2] = t.z; qv[d+3] = t.w;
    }

    // scores for key = lane (always < 49) and key = lane + 32 (valid for lane < 17)
    float s0 = 0.f;
    {
        const float* kr = kh + (size_t)lane * CC;
        #pragma unroll
        for (int d = 0; d < HD; d += 4) {
            const float4 t = *(const float4*)(kr + d);
            s0 += qv[d]*t.x + qv[d+1]*t.y + qv[d+2]*t.z + qv[d+3]*t.w;
        }
    }
    const bool hi = (lane + 32) < KK;
    float s1 = -3.0e38f;
    if (hi) {
        const float* kr = kh + (size_t)(lane + 32) * CC;
        float a = 0.f;
        #pragma unroll
        for (int d = 0; d < HD; d += 4) {
            const float4 t = *(const float4*)(kr + d);
            a += qv[d]*t.x + qv[d+1]*t.y + qv[d+2]*t.z + qv[d+3]*t.w;
        }
        s1 = a;
    }

    // softmax over 49 keys (wave32 shuffle reductions)
    float m = fmaxf(s0, s1);
    #pragma unroll
    for (int off = 16; off; off >>= 1) m = fmaxf(m, __shfl_xor(m, off, 32));
    const float p0 = __expf(s0 - m);
    const float p1 = hi ? __expf(s1 - m) : 0.f;
    float s = p0 + p1;
    #pragma unroll
    for (int off = 16; off; off >>= 1) s += __shfl_xor(s, off, 32);
    const float inv = 1.f / s;

    // A.V: lane -> output dim; v reads are coalesced 128B rows out of L2
    float o = 0.f;
    for (int k = 0; k < KK; ++k) {
        const float pk = __shfl(k < 32 ? p0 : p1, k & 31, 32);
        o += pk * vh[(size_t)k * CC + lane];
    }
    ao[(size_t)bn * CC + (size_t)h * HD + lane] = o * inv;
}

extern "C" void kernel_launch(void* const* d_in, const int* in_sizes, int n_in,
                              void* d_out, int out_size, void* d_ws, size_t ws_size,
                              hipStream_t stream) {
    const float* x   = (const float*)d_in[0];
    const float* xp  = (const float*)d_in[1];
    const int*   idx = (const int*)  d_in[2];
    const float* Wq  = (const float*)d_in[3];
    const float* bq  = (const float*)d_in[4];
    const float* Wk  = (const float*)d_in[5];
    const float* bk  = (const float*)d_in[6];
    const float* Wv  = (const float*)d_in[7];
    const float* bv  = (const float*)d_in[8];
    const float* Wo  = (const float*)d_in[9];
    const float* bo  = (const float*)d_in[10];
    float* out = (float*)d_out;

    const int Mq  = BB * NN;           // 16384
    const int Mkv = BB * WW * KK;      // 35378

    float* qb = (float*)d_ws;
    float* kb = qb + (size_t)Mq  * CC;
    float* vb = kb + (size_t)Mkv * CC;
    float* ab = vb + (size_t)Mkv * CC;

    const float scaling = 0.17677669529663687f;  // HEAD_DIM^-0.5 = 1/sqrt(32)

    auto blocksFor = [](int M) {
        int strips = ((M + 15) / 16) * 4;        // 4 x 64-wide N strips per M tile-row
        return (strips + 7) / 8;                 // 8 waves per 256-thread block
    };

    gemm16_f16<<<blocksFor(Mq),  256, 0, stream>>>(x,  Wq, bq, qb, Mq,  scaling);
    gemm16_f16<<<blocksFor(Mkv), 256, 0, stream>>>(xp, Wk, bk, kb, Mkv, 1.0f);
    gemm16_f16<<<blocksFor(Mkv), 256, 0, stream>>>(xp, Wv, bv, vb, Mkv, 1.0f);
    win_attn  <<<BB * NN,        256, 0, stream>>>(qb, kb, vb, idx, ab);
    gemm16_f16<<<blocksFor(Mq),  256, 0, stream>>>(ab, Wo, bo, out, Mq, 1.0f);
}
